// MultiHeadAttention_798863917312
// MI455X (gfx1250) — matrix operand order repaired and checked by reading, not compiled, not run on hardware
//
#include <hip/hip_runtime.h>
#include <hip/hip_bf16.h>

// CDNA5 / gfx1250 wave32 WMMA types
typedef __attribute__((ext_vector_type(16))) _Float16 v16h;
typedef __attribute__((ext_vector_type(8)))  float    v8f;

#define WMMA_F32_F16(A, B, C) \
  __builtin_amdgcn_wmma_f32_16x16x32_f16(false, (A), false, (B), (short)0, (C), false, false)

// ---------------------------------------------------------------------------
// CDNA5 async global->LDS copy (ASYNCcnt path, ISA 10.x / 15.18.3 op 98).
// One instruction copies 16B per active lane directly into LDS.
// ---------------------------------------------------------------------------
static __device__ inline void async_cp_b128(_Float16* lds, const _Float16* g) {
  const unsigned            loff = (unsigned)(uintptr_t)lds;      // LDS byte offset
  const unsigned long long  ga   = (unsigned long long)(uintptr_t)g;
  asm volatile("global_load_async_to_lds_b128 %0, %1, off"
               :: "v"(loff), "v"(ga) : "memory");
}
static __device__ inline void wait_async0() {
  asm volatile("s_wait_asynccnt 0x0" ::: "memory");
}
static __device__ inline void wave_lds_fence() {
  asm volatile("s_wait_dscnt 0x0" ::: "memory");
}

// ---------------------------------------------------------------------------
// Fragment loaders (wave32 layouts used here; contiguous k order, not the ISA table order)
// A (16x32 f16): lane L row M = L%16, kg = L/16:
//   regs 0..3 K = kg*16+0..7 ; regs 4..7 K = kg*16+8..15
// B (32x16 f16): lane L col N = L%16, kg = L/16: reg i K = kg*16+2i,2i+1
// C/D (16x16 f32): lane L col N = L%16, rows M = (L/16)*8 + v
// ---------------------------------------------------------------------------

static __device__ inline v16h load_A_f32(const float* __restrict__ base, int ld, int lane) {
  const int r = lane & 15, kg = lane >> 4;
  const float* p = base + r * ld + kg * 16;
  v16h a;
#pragma unroll
  for (int i = 0; i < 8; ++i) a[i] = (_Float16)p[i];
#pragma unroll
  for (int i = 0; i < 8; ++i) a[8 + i] = (_Float16)p[8 + i];
  return a;
}

static __device__ inline v16h load_A_f16(const _Float16* base, int ld, int lane) {
  const int r = lane & 15, kg = lane >> 4;
  const _Float16* p = base + r * ld + kg * 16;
  v16h a;
#pragma unroll
  for (int i = 0; i < 8; ++i) a[i] = p[i];
#pragma unroll
  for (int i = 0; i < 8; ++i) a[8 + i] = p[8 + i];
  return a;
}

// B(k,n) = M[n][k]  (B = M^T, M row-major, leading dim ld), f32 source
static __device__ inline v16h load_BT_f32(const float* __restrict__ base, int ld, int lane) {
  const int n = lane & 15, kg = lane >> 4;
  const float* p = base + n * ld + kg * 16;
  v16h b;
#pragma unroll
  for (int i = 0; i < 16; ++i) b[i] = (_Float16)p[i];
  return b;
}

// B(k,n) = M[n][k], f16 source (K^T tiles, contiguous along K)
static __device__ inline v16h load_BT_f16(const _Float16* base, int ld, int lane) {
  const int n = lane & 15, kg = lane >> 4;
  const _Float16* p = base + n * ld + kg * 16;
  v16h b;
#pragma unroll
  for (int i = 0; i < 16; ++i) b[i] = p[i];
  return b;
}

// B(k,n) = M[k][n]  (V tile, row-major, leading dim ld), f16 source
static __device__ inline v16h load_B_f16(const _Float16* base, int ld, int lane) {
  const int n = lane & 15, kg = lane >> 4;
  const _Float16* p = base + (kg * 16) * ld + n;
  v16h b;
#pragma unroll
  for (int i = 0; i < 8; ++i) {
    b[2 * i]     = p[(2 * i) * ld];
    b[2 * i + 1] = p[(2 * i + 1) * ld];
  }
  return b;
}

// ---------------------------------------------------------------------------
// Kernel 1: fused Q/K/V projection.  y = x @ W^T + b, output f16 in [B,H,S,64]
// grid = (512, 32, 3) ; block = 32 (one wave per 16x16 output tile)
// ---------------------------------------------------------------------------
__global__ __launch_bounds__(32)
void qkv_proj_kernel(const float* __restrict__ x,
                     const float* __restrict__ Wq, const float* __restrict__ bq,
                     const float* __restrict__ Wk, const float* __restrict__ bk,
                     const float* __restrict__ Wv, const float* __restrict__ bv,
                     _Float16* __restrict__ Qd, _Float16* __restrict__ Kd,
                     _Float16* __restrict__ Vd)
{
  const int lane = threadIdx.x;
  const int m0 = blockIdx.x * 16;   // row in [0, 8192)
  const int e0 = blockIdx.y * 16;   // col in [0, 512)
  const int z  = blockIdx.z;

  const float* W    = (z == 0) ? Wq : ((z == 1) ? Wk : Wv);
  const float* bias = (z == 0) ? bq : ((z == 1) ? bk : bv);
  _Float16*    outp = (z == 0) ? Qd : ((z == 1) ? Kd : Vd);

  v8f acc = {0.f, 0.f, 0.f, 0.f, 0.f, 0.f, 0.f, 0.f};
#pragma unroll 4
  for (int k0 = 0; k0 < 512; k0 += 32) {
    v16h a = load_A_f32(x + (size_t)m0 * 512 + k0, 512, lane);
    v16h b = load_BT_f32(W + (size_t)e0 * 512 + k0, 512, lane);
    acc = WMMA_F32_F16(a, b, acc);
  }

  const int n    = lane & 15;
  const int half = lane >> 4;
  const float bb = bias[e0 + n];

  const int bidx = m0 >> 12;
  const int s0   = m0 & 4095;
  const int h    = e0 >> 6;
  const int hd0  = e0 & 63;
  _Float16* obase = outp + ((size_t)bidx * 8 + h) * 4096 * 64;

#pragma unroll
  for (int v = 0; v < 8; ++v) {
    const int srow = s0 + half * 8 + v;
    obase[(size_t)srow * 64 + hd0 + n] = (_Float16)(acc[v] + bb);
  }
}

// ---------------------------------------------------------------------------
// Kernel 2: flash attention. 4 waves per block; block handles 64 query rows of
// one (b,h). K/V tiles (32 keys x 64 dims, f16) are staged in LDS once per key
// block via async global->LDS b128 copies and shared by all 4 waves.
// grid = (64, 16) ; block = 128
// ---------------------------------------------------------------------------
#define KV_LD 72   // padded LDS row stride in halves (144B, 16B-aligned rows)

__global__ __launch_bounds__(128)
void flash_attn_kernel(const _Float16* __restrict__ Q,
                       const _Float16* __restrict__ K,
                       const _Float16* __restrict__ V,
                       _Float16* __restrict__ O)   // [B, S, 512] f16
{
  __shared__ __align__(16) _Float16 Kt[32 * KV_LD];   // [key][hd]
  __shared__ __align__(16) _Float16 Vt[32 * KV_LD];   // [key][hd]
  __shared__ __align__(16) _Float16 pb[4][16 * 32];   // per-wave P staging

  const int tid  = threadIdx.x;
  const int lane = tid & 31;
  const int w    = tid >> 5;                  // wave id 0..3
  const int qt   = blockIdx.x * 4 + w;        // query tile 0..255
  const int bh   = blockIdx.y;                // 0..15
  const size_t off = (size_t)bh * 4096 * 64;
  const _Float16* Qb = Q + off;
  const _Float16* Kb = K + off;
  const _Float16* Vb = V + off;
  _Float16* pwave = &pb[w][0];

  // Q fragments stay in registers for the whole pass
  const v16h qa0 = load_A_f16(Qb + (size_t)qt * 16 * 64 + 0,  64, lane);
  const v16h qa1 = load_A_f16(Qb + (size_t)qt * 16 * 64 + 32, 64, lane);

  v8f acc0 = {0.f,0.f,0.f,0.f,0.f,0.f,0.f,0.f};
  v8f acc1 = acc0, acc2 = acc0, acc3 = acc0;
  float mrun[8], lrun[8];
#pragma unroll
  for (int v = 0; v < 8; ++v) { mrun[v] = -1e30f; lrun[v] = 0.0f; }

  const float scale = 0.125f;   // 1/sqrt(64)

  for (int key0 = 0; key0 < 4096; key0 += 32) {
    // ---- stage K and V tiles into LDS (async, 16B chunks, all 128 threads)
    const _Float16* gK = Kb + (size_t)key0 * 64;
    const _Float16* gV = Vb + (size_t)key0 * 64;
#pragma unroll
    for (int it = 0; it < 2; ++it) {
      const int c     = tid + it * 128;   // 0..255 : 32 rows x 8 chunks
      const int row   = c >> 3;
      const int chunk = (c & 7) * 8;      // halves
      async_cp_b128(&Kt[row * KV_LD + chunk], gK + row * 64 + chunk);
      async_cp_b128(&Vt[row * KV_LD + chunk], gV + row * 64 + chunk);
    }
    wait_async0();
    __syncthreads();

    // ---- scores S = Q @ K^T (two 16-key N-tiles, K-dim 64 = 2 wmma steps)
    v8f s0 = {0.f,0.f,0.f,0.f,0.f,0.f,0.f,0.f};
    v8f s1 = s0;
    s0 = WMMA_F32_F16(qa0, load_BT_f16(Kt + 0 * KV_LD + 0,  KV_LD, lane), s0);
    s0 = WMMA_F32_F16(qa1, load_BT_f16(Kt + 0 * KV_LD + 32, KV_LD, lane), s0);
    s1 = WMMA_F32_F16(qa0, load_BT_f16(Kt + 16 * KV_LD + 0,  KV_LD, lane), s1);
    s1 = WMMA_F32_F16(qa1, load_BT_f16(Kt + 16 * KV_LD + 32, KV_LD, lane), s1);

    // ---- online softmax (row reductions across 16 column-lanes per half)
    float p0[8], p1[8], alpha[8];
#pragma unroll
    for (int v = 0; v < 8; ++v) {
      const float a = s0[v] * scale;
      const float b = s1[v] * scale;
      float t = fmaxf(a, b);
#pragma unroll
      for (int mask = 1; mask < 16; mask <<= 1)
        t = fmaxf(t, __shfl_xor(t, mask, 32));
      const float mnew = fmaxf(mrun[v], t);
      alpha[v] = __expf(mrun[v] - mnew);
      p0[v] = __expf(a - mnew);
      p1[v] = __expf(b - mnew);
      float r = p0[v] + p1[v];
#pragma unroll
      for (int mask = 1; mask < 16; mask <<= 1)
        r += __shfl_xor(r, mask, 32);
      lrun[v] = lrun[v] * alpha[v] + r;
      mrun[v] = mnew;
    }
#pragma unroll
    for (int v = 0; v < 8; ++v) {
      acc0[v] *= alpha[v]; acc1[v] *= alpha[v];
      acc2[v] *= alpha[v]; acc3[v] *= alpha[v];
    }

    // ---- P: C-layout regs -> per-wave LDS -> A-layout regs (wave-local)
    {
      const int n = lane & 15, half = lane >> 4;
#pragma unroll
      for (int v = 0; v < 8; ++v) {
        const int row = half * 8 + v;
        pwave[row * 32 + n]      = (_Float16)p0[v];
        pwave[row * 32 + 16 + n] = (_Float16)p1[v];
      }
    }
    wave_lds_fence();
    const v16h pa = load_A_f16(pwave, 32, lane);

    // ---- O += P @ V  (K-dim = 32 keys, N = 64 head dims = 4 tiles)
    acc0 = WMMA_F32_F16(pa, load_B_f16(Vt + 0,  KV_LD, lane), acc0);
    acc1 = WMMA_F32_F16(pa, load_B_f16(Vt + 16, KV_LD, lane), acc1);
    acc2 = WMMA_F32_F16(pa, load_B_f16(Vt + 32, KV_LD, lane), acc2);
    acc3 = WMMA_F32_F16(pa, load_B_f16(Vt + 48, KV_LD, lane), acc3);

    __syncthreads();   // protect Kt/Vt before next iteration's staging
  }

  // ---- normalize and store merged-head output O[b, s, h*64 + hd] (f16)
  const int bidx = bh >> 3, h = bh & 7;
  const int n = lane & 15, half = lane >> 4;
#pragma unroll
  for (int v = 0; v < 8; ++v) {
    const float inv = 1.0f / lrun[v];
    const int srow = qt * 16 + half * 8 + v;
    const size_t o = ((size_t)bidx * 4096 + srow) * 512 + (size_t)h * 64;
    O[o +  0 + n] = (_Float16)(acc0[v] * inv);
    O[o + 16 + n] = (_Float16)(acc1[v] * inv);
    O[o + 32 + n] = (_Float16)(acc2[v] * inv);
    O[o + 48 + n] = (_Float16)(acc3[v] * inv);
  }
}

// ---------------------------------------------------------------------------
// Kernel 3: output projection. y = attn @ Wo^T + bo  -> f32 d_out
// grid = (512, 32) ; block = 32
// ---------------------------------------------------------------------------
__global__ __launch_bounds__(32)
void out_proj_kernel(const _Float16* __restrict__ A,   // [8192, 512] f16
                     const float* __restrict__ Wo, const float* __restrict__ bo,
                     float* __restrict__ Y)            // [8192, 512] f32
{
  const int lane = threadIdx.x;
  const int m0 = blockIdx.x * 16;
  const int e0 = blockIdx.y * 16;

  v8f acc = {0.f,0.f,0.f,0.f,0.f,0.f,0.f,0.f};
#pragma unroll 4
  for (int k0 = 0; k0 < 512; k0 += 32) {
    v16h a = load_A_f16(A + (size_t)m0 * 512 + k0, 512, lane);
    v16h b = load_BT_f32(Wo + (size_t)e0 * 512 + k0, 512, lane);
    acc = WMMA_F32_F16(a, b, acc);
  }

  const int n = lane & 15, half = lane >> 4;
  const float bb = bo[e0 + n];
#pragma unroll
  for (int v = 0; v < 8; ++v) {
    const int row = m0 + half * 8 + v;
    Y[(size_t)row * 512 + e0 + n] = acc[v] + bb;
  }
}

// ---------------------------------------------------------------------------
extern "C" void kernel_launch(void* const* d_in, const int* in_sizes, int n_in,
                              void* d_out, int out_size, void* d_ws, size_t ws_size,
                              hipStream_t stream) {
  (void)in_sizes; (void)n_in; (void)out_size; (void)ws_size;

  const float* x  = (const float*)d_in[0];
  const float* Wq = (const float*)d_in[1];
  const float* bq = (const float*)d_in[2];
  const float* Wk = (const float*)d_in[3];
  const float* bk = (const float*)d_in[4];
  const float* Wv = (const float*)d_in[5];
  const float* bv = (const float*)d_in[6];
  const float* Wo = (const float*)d_in[7];
  const float* bo = (const float*)d_in[8];
  float* out = (float*)d_out;

  // workspace: Q, K, V in [B,H,S,64] f16; attn output in [B,S,512] f16 (32 MB)
  const size_t elems = (size_t)2 * 8 * 4096 * 64;   // 4,194,304
  _Float16* Qd = (_Float16*)d_ws;
  _Float16* Kd = Qd + elems;
  _Float16* Vd = Kd + elems;
  _Float16* Od = Vd + elems;

  qkv_proj_kernel<<<dim3(512, 32, 3), dim3(32), 0, stream>>>(
      x, Wq, bq, Wk, bk, Wv, bv, Qd, Kd, Vd);
  flash_attn_kernel<<<dim3(64, 16), dim3(128), 0, stream>>>(Qd, Kd, Vd, Od);
  out_proj_kernel<<<dim3(512, 32), dim3(32), 0, stream>>>(Od, Wo, bo, out);
}
